// TransformerDecoder_51462298141143
// MI455X (gfx1250) — matrix operand order repaired and checked by reading, not compiled, not run on hardware
//
#include <hip/hip_runtime.h>
#include <hip/hip_bf16.h>

typedef __attribute__((ext_vector_type(16))) _Float16 v16h;
typedef __attribute__((ext_vector_type(8)))  _Float16 v8h;
typedef __attribute__((ext_vector_type(8)))  float    v8f;

#define DEVI __device__ __forceinline__

// ---------------------------------------------------------------------------
// Elementwise f32 -> f16 convert
// ---------------------------------------------------------------------------
__global__ void k_f32_to_f16(const float* __restrict__ x, _Float16* __restrict__ y, int n) {
    int i = blockIdx.x * blockDim.x + threadIdx.x;
    if (i < n) y[i] = (_Float16)x[i];
}

// ---------------------------------------------------------------------------
// Weight transpose + convert: W (K x N) f32  ->  Wt (N x K) f16
// ---------------------------------------------------------------------------
__global__ void k_wT(const float* __restrict__ W, _Float16* __restrict__ Wt,
                     int K, int N) {
    long long i = (long long)blockIdx.x * blockDim.x + threadIdx.x;
    long long total = (long long)K * N;
    if (i >= total) return;
    int n = (int)(i / K);
    int k = (int)(i % K);
    Wt[i] = (_Float16)W[(long long)k * N + n];
}

// ---------------------------------------------------------------------------
// v_wmma_f32_16x16x32_f16 fragment helpers (wave32)
// A (16x32): lanes 0-15 row=lane, halves [K0..15];
//            lanes 16-31 row=lane-16, halves [K16..31].
// B from Wt (N x K): col=lane&15, kb=(lane>>4)*16, 16 contiguous halves.
// ---------------------------------------------------------------------------
DEVI v16h load_a16(const _Float16* __restrict__ pa, int k0) {
    v8h lo = *(const v8h*)(pa + k0);
    v8h hi = *(const v8h*)(pa + k0 + 8);
    return __builtin_shufflevector(lo, hi,
               0, 1, 2, 3, 4, 5, 6, 7, 8, 9, 10, 11, 12, 13, 14, 15);
}

template<bool RELU_IN>
DEVI v16h relu_frag(v16h a) {
    if (RELU_IN) {
#pragma unroll
        for (int i = 0; i < 16; ++i)
            a[i] = (a[i] > (_Float16)0.f) ? a[i] : (_Float16)0.f;
    }
    return a;
}

DEVI v8f wmma32(v16h a, v16h b, v8f c) {
    return __builtin_amdgcn_wmma_f32_16x16x32_f16(false, a, false, b,
                                                  (short)0, c, false, false);
}

template<bool RELU_OUT>
DEVI void store_tile(v8f c, const float* __restrict__ bias,
                     _Float16* __restrict__ O16, float* __restrict__ O32,
                     int rowS, int col, int N) {
    float bv = bias ? bias[col] : 0.f;
#pragma unroll
    for (int r = 0; r < 8; ++r) {
        float x = c[r] + bv;
        if (RELU_OUT) x = fmaxf(x, 0.f);
        size_t o = (size_t)(rowS + r) * N + col;
        if (O16) O16[o] = (_Float16)x;
        if (O32) O32[o] = x;
    }
}

// ---------------------------------------------------------------------------
// Fused GEMM: out = act(A[MxK] @ W[KxN] + bias), weights pre-transposed (NxK).
// Block = 256 threads = 8 waves; block tile 256(M) x 16*TILES(N).
// Each wave owns 32 rows x TILES*16 cols: 2 A frags + TILES B frags ->
// 2*TILES WMMAs per K-chunk. K loop is unrolled x2 with ping-pong fragment
// buffers (no rotate copies); loads for chunk c+1 issue before the WMMAs of
// chunk c. Branch-free body (TILES is a template int); masked tail handles
// K%32 (only K=48). Requires M % 32 == 0 (32 / 2048 / 131072 here); all
// guards wave-uniform => EXEC all-ones for WMMA.
// ---------------------------------------------------------------------------
template<int TILES, bool RELU_OUT, bool RELU_IN>
__global__ void __launch_bounds__(256)
k_gemm(const _Float16* __restrict__ A, const _Float16* __restrict__ Wt,
       const float* __restrict__ bias,
       _Float16* __restrict__ O16, float* __restrict__ O32,
       int M, int N, int K) {
    const int lane = threadIdx.x & 31;
    const int wave = threadIdx.x >> 5;
    const int row0 = blockIdx.y * 256 + wave * 32;   // wave covers 32 rows
    const int colB = blockIdx.x * (16 * TILES);
    if (row0 >= M) return;                            // wave-uniform

    const int lh = lane >> 4;                         // 0 / 1
    const int lm = lane & 15;
    const _Float16* pa0 = A + (size_t)(row0 + lm) * K + lh * 16;
    const _Float16* pa1 = pa0 + (size_t)16 * K;
    const _Float16* pb0 = Wt + (size_t)(colB + lm) * K + lh * 16;
    const size_t ts = (size_t)16 * K;                 // B col-tile stride

    v8f vz = {};
    v8f acc0[TILES], acc1[TILES];
#pragma unroll
    for (int t = 0; t < TILES; ++t) { acc0[t] = vz; acc1[t] = vz; }

    // ping-pong fragment buffers (no rotate copies)
    v16h a0A, a1A, bA[TILES];
    v16h a0B, a1B, bB[TILES];

    auto load_chunk = [&](v16h& a0, v16h& a1, v16h (&bf)[TILES], int k0) {
        a0 = load_a16(pa0, k0);
        a1 = load_a16(pa1, k0);
#pragma unroll
        for (int t = 0; t < TILES; ++t)
            bf[t] = *(const v16h*)(pb0 + t * ts + k0);
    };
    auto compute_chunk = [&](v16h a0, v16h a1, v16h (&bf)[TILES]) {
        v16h ra0 = relu_frag<RELU_IN>(a0);
        v16h ra1 = relu_frag<RELU_IN>(a1);
#pragma unroll
        for (int t = 0; t < TILES; ++t) {
            acc0[t] = wmma32(ra0, bf[t], acc0[t]);
            acc1[t] = wmma32(ra1, bf[t], acc1[t]);
        }
    };

    const int kmain = K & ~31;
    const int nch = kmain >> 5;                        // # full 32-K chunks (>=1)

    load_chunk(a0A, a1A, bA, 0);                       // chunk 0 -> buf A
    int c = 1;
    for (; c + 1 < nch; c += 2) {
        load_chunk(a0B, a1B, bB, c * 32);              // odd chunk -> buf B
        compute_chunk(a0A, a1A, bA);
        load_chunk(a0A, a1A, bA, (c + 1) * 32);        // even chunk -> buf A
        compute_chunk(a0B, a1B, bB);
    }
    if (c < nch) {                                     // one trailing chunk
        load_chunk(a0B, a1B, bB, c * 32);
        compute_chunk(a0A, a1A, bA);
        compute_chunk(a0B, a1B, bB);
    } else {
        compute_chunk(a0A, a1A, bA);
    }

    // ---- masked K tail (only K=48 reaches this) ----
    if (kmain < K) {
        const int k0 = kmain;
        v16h a0, a1;
#pragma unroll
        for (int i = 0; i < 8; ++i) {
            int k1 = k0 + lh * 16 + i;
            int k2 = k1 + 8;
            a0[i]     = (k1 < K) ? pa0[k0 + i]      : (_Float16)0.f;
            a0[8 + i] = (k2 < K) ? pa0[k0 + 8 + i] : (_Float16)0.f;
            a1[i]     = (k1 < K) ? pa1[k0 + i]      : (_Float16)0.f;
            a1[8 + i] = (k2 < K) ? pa1[k0 + 8 + i] : (_Float16)0.f;
        }
        a0 = relu_frag<RELU_IN>(a0);
        a1 = relu_frag<RELU_IN>(a1);
#pragma unroll
        for (int t = 0; t < TILES; ++t) {
            v16h b;
#pragma unroll
            for (int j = 0; j < 16; ++j) {
                int k = k0 + lh * 16 + j;
                b[j] = (k < K) ? pb0[t * ts + k0 + j] : (_Float16)0.f;
            }
            acc0[t] = wmma32(a0, b, acc0[t]);
            acc1[t] = wmma32(a1, b, acc1[t]);
        }
    }

    // ---- store ----
    const int rowS0 = row0 + lh * 8;
    const int rowS1 = rowS0 + 16;
#pragma unroll
    for (int t = 0; t < TILES; ++t) {
        store_tile<RELU_OUT>(acc0[t], bias, O16, O32, rowS0, colB + t * 16 + lm, N);
        store_tile<RELU_OUT>(acc1[t], bias, O16, O32, rowS1, colB + t * 16 + lm, N);
    }
}

// ---------------------------------------------------------------------------
// new_edges = att*em + att + ea,  att = q_i*k_j/sqrt(H); per-head (24) sums
// ---------------------------------------------------------------------------
__global__ void __launch_bounds__(192)
k_edge_att(const _Float16* __restrict__ q, const _Float16* __restrict__ k,
           const _Float16* __restrict__ em, const _Float16* __restrict__ ea,
           _Float16* __restrict__ ne, float* __restrict__ logits) {
    int e = blockIdx.x;                 // b*4096 + i*64 + j
    int b = e >> 12;
    int i = (e >> 6) & 63;
    int j = e & 63;
    int t = threadIdx.x;                // 0..191
    float qv = (float)q[((long long)(b * 64 + i)) * 192 + t];
    float kv = (float)k[((long long)(b * 64 + j)) * 192 + t];
    float att = qv * kv * 0.0721687836f;    // 1/sqrt(192)
    long long o = (long long)e * 192 + t;
    float nev = att * (float)em[o] + att + (float)ea[o];
    ne[o] = (_Float16)nev;
    __shared__ float s[192];
    s[t] = nev;
    __syncthreads();
    if (t < 8) {
        float sum = 0.f;
#pragma unroll
        for (int d = 0; d < 24; ++d) sum += s[t * 24 + d];
        logits[(long long)e * 8 + t] = sum;
    }
}

// ---------------------------------------------------------------------------
// Softmax over j (64 keys) per (b,i,head); one wave per head, in-place.
// ---------------------------------------------------------------------------
__global__ void __launch_bounds__(256)
k_softmax(float* __restrict__ att) {
    int bi = blockIdx.x;                        // b*64 + i
    int h = threadIdx.x >> 5, lane = threadIdx.x & 31;
    float* base = att + (long long)bi * 512 + h;    // stride 8 over j
    float x0 = base[lane * 8];
    float x1 = base[(lane + 32) * 8];
    float mx = fmaxf(x0, x1);
#pragma unroll
    for (int o = 16; o > 0; o >>= 1) mx = fmaxf(mx, __shfl_xor(mx, o, 32));
    float e0 = __expf(x0 - mx), e1 = __expf(x1 - mx);
    float s = e0 + e1;
#pragma unroll
    for (int o = 16; o > 0; o >>= 1) s += __shfl_xor(s, o, 32);
    float inv = 1.f / s;
    base[lane * 8] = e0 * inv;
    base[(lane + 32) * 8] = e1 * inv;
}

// ---------------------------------------------------------------------------
// wv[b,i,h,d] = sum_j attn[b,i,j,h] * v[b,j,h*24+d]
// ---------------------------------------------------------------------------
__global__ void __launch_bounds__(192)
k_wv(const float* __restrict__ att, const _Float16* __restrict__ v,
     _Float16* __restrict__ wv) {
    int bi = blockIdx.x;                 // b*64 + i
    int b = bi >> 6;
    int t = threadIdx.x;                 // h*24 + d
    int h = t / 24;
    const float* a = att + (long long)bi * 512 + h;
    float acc = 0.f;
#pragma unroll 8
    for (int j = 0; j < 64; ++j)
        acc += a[j * 8] * (float)v[((long long)(b * 64 + j)) * 192 + t];
    wv[(long long)bi * 192 + t] = (_Float16)acc;
}

// ---------------------------------------------------------------------------
// Fused residual-add + LayerNorm over D=192; one wave per row (6 elems/lane).
// ---------------------------------------------------------------------------
__global__ void __launch_bounds__(256)
k_add_ln(const _Float16* __restrict__ x, const _Float16* __restrict__ r,
         const float* __restrict__ g, const float* __restrict__ b,
         _Float16* __restrict__ out, int rows) {
    int wave = threadIdx.x >> 5, lane = threadIdx.x & 31;
    int row = blockIdx.x * 8 + wave;
    if (row >= rows) return;
    const _Float16* px = x + (long long)row * 192;
    const _Float16* pr = r + (long long)row * 192;
    float v[6]; float s = 0.f;
#pragma unroll
    for (int i = 0; i < 6; ++i) {
        v[i] = (float)px[lane + 32 * i] + (float)pr[lane + 32 * i];
        s += v[i];
    }
#pragma unroll
    for (int o = 16; o > 0; o >>= 1) s += __shfl_xor(s, o, 32);
    float m = s * (1.f / 192.f);
    float qq = 0.f;
#pragma unroll
    for (int i = 0; i < 6; ++i) { float d = v[i] - m; qq += d * d; }
#pragma unroll
    for (int o = 16; o > 0; o >>= 1) qq += __shfl_xor(qq, o, 32);
    float inv = rsqrtf(qq * (1.f / 192.f) + 1e-5f);
    _Float16* po = out + (long long)row * 192;
#pragma unroll
    for (int i = 0; i < 6; ++i) {
        int c = lane + 32 * i;
        po[c] = (_Float16)((v[i] - m) * inv * g[c] + b[c]);
    }
}

// ---------------------------------------------------------------------------
// Host orchestration. Input order = JAX pytree flatten (alphabetical keys):
//   0: latents
//   1..4: create_edges   5..8: create_nodes   9..12: edge_tf
//   13..132: layers x4 (30 arrays each):
//     +0 add.b +1 add.w +2 edges_out.b +3 edges_out.w +4 k.b +5 k.w
//     +6 ln_e.b +7 ln_e.g +8 ln_e2.b +9 ln_e2.g +10 ln_n.b +11 ln_n.g
//     +12 ln_n2.b +13 ln_n2.g +14..17 mlp_e{b,w,b,w} +18..21 mlp_n{b,w,b,w}
//     +22 mul.b +23 mul.w +24 nodes_out.b +25 nodes_out.w +26 q.b +27 q.w
//     +28 v.b +29 v.w
//   133..136: node_tf   137..140: out_edges   141..144: out_nodes
// ---------------------------------------------------------------------------
extern "C" void kernel_launch(void* const* d_in, const int* in_sizes, int n_in,
                              void* d_out, int out_size, void* d_ws, size_t ws_size,
                              hipStream_t stream) {
    (void)in_sizes; (void)n_in; (void)out_size; (void)ws_size;
    auto F = [&](int i) { return (const float*)d_in[i]; };

    const int B = 32, N = 64, ND = 48, ED = 32, G = 256, H = 192;
    const int BN = B * N;           // 2048 node rows
    const int BNN = B * N * N;      // 131072 edge rows

    char* ws = (char*)d_ws;
    size_t off = 0;
    auto alloc = [&](size_t bytes) -> void* {
        void* p = ws + off;
        off += (bytes + 255) & ~(size_t)255;
        return p;
    };
    auto allocH = [&](size_t e) { return (_Float16*)alloc(e * sizeof(_Float16)); };
    auto allocF = [&](size_t e) { return (float*)alloc(e * sizeof(float)); };

    // TILES=4 when N%64==0 (all big GEMMs); 3 for N=48; 2 for N=32.
    auto gemm = [&](const _Float16* A, const _Float16* Wt, const float* bias,
                    _Float16* O16, float* O32, int M, int Nout, int K,
                    bool relu_out, bool relu_in) {
        dim3 blk(256);
        if ((Nout & 63) == 0) {
            dim3 grid(Nout / 64, (M + 255) / 256);
            if (relu_out) {
                if (relu_in) k_gemm<4, true,  true ><<<grid, blk, 0, stream>>>(A, Wt, bias, O16, O32, M, Nout, K);
                else         k_gemm<4, true,  false><<<grid, blk, 0, stream>>>(A, Wt, bias, O16, O32, M, Nout, K);
            } else {
                if (relu_in) k_gemm<4, false, true ><<<grid, blk, 0, stream>>>(A, Wt, bias, O16, O32, M, Nout, K);
                else         k_gemm<4, false, false><<<grid, blk, 0, stream>>>(A, Wt, bias, O16, O32, M, Nout, K);
            }
        } else if (Nout == 48) {
            dim3 grid(1, (M + 255) / 256);
            k_gemm<3, false, false><<<grid, blk, 0, stream>>>(A, Wt, bias, O16, O32, M, Nout, K);
        } else {  // Nout == 32
            dim3 grid(1, (M + 255) / 256);
            k_gemm<2, false, false><<<grid, blk, 0, stream>>>(A, Wt, bias, O16, O32, M, Nout, K);
        }
    };
    auto prepW = [&](int idx, int K, int Nout) -> _Float16* {
        _Float16* wt = allocH((size_t)K * Nout);
        long long total = (long long)K * Nout;
        k_wT<<<(int)((total + 255) / 256), 256, 0, stream>>>(F(idx), wt, K, Nout);
        return wt;
    };

    // ---- persistent activation buffers (f16 halves HBM traffic) ----
    _Float16* lat16   = allocH((size_t)B * G);
    _Float16* tceA    = allocH((size_t)B * G);           // (32,256)
    _Float16* edges32 = allocH((size_t)BNN * ED);        // view of (32, N*N*ED)
    _Float16* tcn     = allocH((size_t)B * N * ND);      // (32,3072)
    _Float16* nodes48 = allocH((size_t)B * N * ND);      // view (2048,48)
    _Float16* nodes_h = allocH((size_t)BN * H);
    _Float16* bq      = allocH((size_t)BN * H);
    _Float16* bk      = allocH((size_t)BN * H);
    _Float16* bv      = allocH((size_t)BN * H);
    _Float16* bwv     = allocH((size_t)BN * H);
    _Float16* ban     = allocH((size_t)BN * H);
    _Float16* btn     = allocH((size_t)BN * H);
    _Float16* edges_h = allocH((size_t)BNN * H);         // 50 MB each
    _Float16* EM      = allocH((size_t)BNN * H);
    _Float16* EA      = allocH((size_t)BNN * H);
    _Float16* NE      = allocH((size_t)BNN * H);
    float*    logits  = allocF((size_t)BNN * 8);

    size_t mark = off;   // transient weight region; stream order makes reuse safe

    k_f32_to_f16<<<(B * G + 255) / 256, 256, 0, stream>>>(F(0), lat16, B * G);

    // ---- create_edges ----
    { _Float16* w = prepW(2, G, G);
      gemm(lat16, w, F(1), tceA, nullptr, B, G, G, true, false); off = mark; }
    { _Float16* w = prepW(4, G, N * N * ED);
      gemm(tceA, w, F(3), edges32, nullptr, B, N * N * ED, G, true, false); off = mark; }

    // ---- create_nodes ----
    { _Float16* w = prepW(6, G, N * ND);
      gemm(lat16, w, F(5), tcn, nullptr, B, N * ND, G, true, false); off = mark; }
    { _Float16* w = prepW(8, N * ND, N * ND);
      gemm(tcn, w, F(7), nodes48, nullptr, B, N * ND, N * ND, true, false); off = mark; }

    // ---- node_tf: (2048,48)->(2048,192)->(2048,192) ----
    { _Float16* w = prepW(134, ND, H);
      gemm(nodes48, w, F(133), btn, nullptr, BN, H, ND, true, false); off = mark; }
    { _Float16* w = prepW(136, H, H);
      gemm(btn, w, F(135), nodes_h, nullptr, BN, H, H, true, false); off = mark; }

    // ---- edge_tf: (131072,32)->(131072,192)->(131072,192) ----
    { _Float16* w = prepW(10, ED, H);
      gemm(edges32, w, F(9), EM, nullptr, BNN, H, ED, true, false); off = mark; }
    { _Float16* w = prepW(12, H, H);
      gemm(EM, w, F(11), edges_h, nullptr, BNN, H, H, true, false); off = mark; }

    // ---- 4 transformer layers ----
    for (int l = 0; l < 4; ++l) {
        int pb = 13 + 30 * l;
        _Float16* Wq  = prepW(pb + 27, H, H);
        _Float16* Wk  = prepW(pb + 5,  H, H);
        _Float16* Wv  = prepW(pb + 29, H, H);
        _Float16* Wm  = prepW(pb + 23, H, H);
        _Float16* Wa  = prepW(pb + 1,  H, H);
        _Float16* Wno = prepW(pb + 25, H, H);
        _Float16* Weo = prepW(pb + 3,  H, H);
        _Float16* Wn0 = prepW(pb + 19, H, H);
        _Float16* Wn1 = prepW(pb + 21, H, H);
        _Float16* We0 = prepW(pb + 15, H, H);
        _Float16* We1 = prepW(pb + 17, H, H);

        gemm(nodes_h, Wq, F(pb + 26), bq, nullptr, BN, H, H, false, false);
        gemm(nodes_h, Wk, F(pb + 4),  bk, nullptr, BN, H, H, false, false);
        gemm(nodes_h, Wv, F(pb + 28), bv, nullptr, BN, H, H, false, false);
        gemm(edges_h, Wm, F(pb + 22), EM, nullptr, BNN, H, H, false, false);
        gemm(edges_h, Wa, F(pb + 0),  EA, nullptr, BNN, H, H, false, false);

        k_edge_att<<<BNN, H, 0, stream>>>(bq, bk, EM, EA, NE, logits);
        k_softmax<<<BN, 256, 0, stream>>>(logits);
        k_wv<<<BN, H, 0, stream>>>(logits, bv, bwv);

        gemm(bwv, Wno, F(pb + 24), ban, nullptr, BN, H, H, false, false);
        gemm(NE, Weo, F(pb + 2), EM, nullptr, BNN, H, H, false, true);  // relu on input

        k_add_ln<<<BN / 8, 256, 0, stream>>>(ban, nodes_h, F(pb + 11), F(pb + 10), nodes_h, BN);   // ln_n
        k_add_ln<<<BNN / 8, 256, 0, stream>>>(EM, edges_h, F(pb + 7),  F(pb + 6),  edges_h, BNN);  // ln_e

        gemm(nodes_h, Wn0, F(pb + 18), btn, nullptr, BN, H, H, true, false);
        gemm(btn,     Wn1, F(pb + 20), ban, nullptr, BN, H, H, false, false);
        k_add_ln<<<BN / 8, 256, 0, stream>>>(ban, nodes_h, F(pb + 13), F(pb + 12), nodes_h, BN);   // ln_n2

        gemm(edges_h, We0, F(pb + 14), EA, nullptr, BNN, H, H, true, false);
        gemm(EA,      We1, F(pb + 16), EM, nullptr, BNN, H, H, false, false);
        k_add_ln<<<BNN / 8, 256, 0, stream>>>(EM, edges_h, F(pb + 9), F(pb + 8), edges_h, BNN);    // ln_e2

        off = mark;
    }

    // ---- output heads (f32 straight to d_out: nodes then edges) ----
    float* out_nodes_f = (float*)d_out;
    float* out_edges_f = (float*)d_out + (size_t)BN * ND;

    { _Float16* w = prepW(142, H, H);
      gemm(nodes_h, w, F(141), btn, nullptr, BN, H, H, true, false); off = mark; }
    { _Float16* w = prepW(144, H, ND);
      gemm(btn, w, F(143), nullptr, out_nodes_f, BN, ND, H, false, false); off = mark; }
    { _Float16* w = prepW(138, H, H);
      gemm(edges_h, w, F(137), EM, nullptr, BNN, H, H, true, false); off = mark; }
    { _Float16* w = prepW(140, H, ED);
      gemm(EM, w, F(139), nullptr, out_edges_f, BNN, ED, H, false, false); off = mark; }
}
